// MultiheadSelfAttention_14577119003257
// MI455X (gfx1250) — matrix operand order repaired and checked by reading, not compiled, not run on hardware
//
#include <hip/hip_runtime.h>

#define BB 4
#define TT 2048
#define DD 1024
#define HH 16
#define HD 64

typedef __attribute__((ext_vector_type(16))) __bf16          bf16x16;
typedef __attribute__((ext_vector_type(8)))  float           f32x8;
typedef __attribute__((ext_vector_type(16))) unsigned short  u16x16;
typedef __attribute__((ext_vector_type(8)))  unsigned short  u16x8;

// float -> bf16 bits, round-to-nearest-even
__device__ __forceinline__ unsigned short f2bf(float f) {
  unsigned u = __builtin_bit_cast(unsigned, f);
  u += 0x7FFFu + ((u >> 16) & 1u);
  return (unsigned short)(u >> 16);
}

__device__ __forceinline__ f32x8 wmma_bf16(u16x16 a, u16x16 b, f32x8 c) {
  return __builtin_amdgcn_wmma_f32_16x16x32_bf16(
      false, __builtin_bit_cast(bf16x16, a),
      false, __builtin_bit_cast(bf16x16, b),
      (short)0, c, false, false);
}

// A-fragment: 16 rows x 32 K, row-major source, lane<16: row=lane, K={0..7,16..23};
// lane>=16: row=lane-16, K={8..15,24..31}   (ISA 7.12.2, 16-bit A 16x32)
__device__ __forceinline__ u16x16 load_A(const unsigned short* p, int stride) {
  const int lane = threadIdx.x & 31;
  const unsigned short* q = p + (lane & 15) * stride + ((lane >> 4) << 3);
  u16x8 lo = *(const u16x8*)(q);
  u16x8 hi = *(const u16x8*)(q + 16);
  u16x16 r;
#pragma unroll
  for (int i = 0; i < 8; ++i) { r[i] = lo[i]; r[i + 8] = hi[i]; }
  return r;
}

// B-fragment: K=32 x 16 cols; source is the N x K (row-major) operand, so lane reads
// row n = column n of B. lane<16: K={0..7,16..23}, lane>=16: K={8..15,24..31}.
__device__ __forceinline__ u16x16 load_B(const unsigned short* p, int stride) {
  const int lane = threadIdx.x & 31;
  const unsigned short* q = p + (lane & 15) * stride + ((lane >> 4) << 3);
  u16x8 lo = *(const u16x8*)(q);
  u16x8 hi = *(const u16x8*)(q + 16);
  u16x16 r;
#pragma unroll
  for (int i = 0; i < 8; ++i) { r[i] = lo[i]; r[i + 8] = hi[i]; }
  return r;
}

__global__ void cvt_bf16(const float* __restrict__ in, unsigned short* __restrict__ out, int n) {
  int i = blockIdx.x * blockDim.x + threadIdx.x;
  if (i < n) out[i] = f2bf(in[i]);
}

// ---------------------------------------------------------------------------
// qkv = x @ W_qkv^T ; scatter into q[B,H,T,hd]*scale, k[B,H,T,hd], v^T[B,H,hd,T]
// Wave tile: 16 rows x 64 cols. Ping-pong double buffering (2x-unrolled K loop):
// loads write directly into the alternate buffer, so no rotation moves.
// ---------------------------------------------------------------------------
__global__ void __launch_bounds__(256) qkv_gemm(
    const unsigned short* __restrict__ xb,
    const unsigned short* __restrict__ wb,
    unsigned short* __restrict__ qd,
    unsigned short* __restrict__ kd,
    unsigned short* __restrict__ vtd) {
  const int lane = threadIdx.x & 31;
  const int wave = blockIdx.x * (blockDim.x >> 5) + (threadIdx.x >> 5);
  const int nstrip = wave % 48;   // 48 strips of 64 cols over 3072
  const int mtile  = wave / 48;   // 512 tiles of 16 rows over B*T
  const unsigned short* ap = xb + (size_t)mtile * 16 * DD;
  const unsigned short* bp = wb + (size_t)(nstrip * 64) * DD;

  f32x8 acc[4] = {};
  u16x16 a0, a1, b0[4], b1[4];

  a0 = load_A(ap, DD);
#pragma unroll
  for (int jt = 0; jt < 4; ++jt) b0[jt] = load_B(bp + (size_t)jt * 16 * DD, DD);

  for (int kk = 0; kk < DD - 64; kk += 64) {
    a1 = load_A(ap + kk + 32, DD);
#pragma unroll
    for (int jt = 0; jt < 4; ++jt) b1[jt] = load_B(bp + (size_t)jt * 16 * DD + kk + 32, DD);
#pragma unroll
    for (int jt = 0; jt < 4; ++jt) acc[jt] = wmma_bf16(a0, b0[jt], acc[jt]);

    a0 = load_A(ap + kk + 64, DD);
#pragma unroll
    for (int jt = 0; jt < 4; ++jt) b0[jt] = load_B(bp + (size_t)jt * 16 * DD + kk + 64, DD);
#pragma unroll
    for (int jt = 0; jt < 4; ++jt) acc[jt] = wmma_bf16(a1, b1[jt], acc[jt]);
  }
  a1 = load_A(ap + DD - 32, DD);
#pragma unroll
  for (int jt = 0; jt < 4; ++jt) b1[jt] = load_B(bp + (size_t)jt * 16 * DD + DD - 32, DD);
#pragma unroll
  for (int jt = 0; jt < 4; ++jt) acc[jt] = wmma_bf16(a0, b0[jt], acc[jt]);
#pragma unroll
  for (int jt = 0; jt < 4; ++jt) acc[jt] = wmma_bf16(a1, b1[jt], acc[jt]);

  const int col  = lane & 15;
  const int qb16 = (lane >> 4) << 3;
#pragma unroll
  for (int jt = 0; jt < 4; ++jt) {
    const int e = nstrip * 64 + jt * 16 + col;
    const int which = e >> 10;           // 0=q 1=k 2=v
    const int h = (e & 1023) >> 6;
    const int d = e & 63;
#pragma unroll
    for (int r = 0; r < 8; ++r) {
      const int grow = mtile * 16 + qb16 + r;
      const int bidx = grow >> 11;
      const int t = grow & (TT - 1);
      const size_t bh = (size_t)(bidx * HH + h);
      const float v = acc[jt][r];
      if (which == 0)      qd[(bh * TT + t) * HD + d] = f2bf(v * 0.125f);
      else if (which == 1) kd[(bh * TT + t) * HD + d] = f2bf(v);
      else                 vtd[(bh * HD + d) * TT + t] = f2bf(v);
    }
  }
}

// ---------------------------------------------------------------------------
// Flash attention: one wave per (b,h,16-query tile). Transposed score tiles so
// the exp'd S^T C-fragments are directly the A-fragment of P @ V.
// Per chunk: load K, load V (in flight during score WMMAs), softmax hides V.
// ---------------------------------------------------------------------------
__global__ void __launch_bounds__(256) attn_kernel(
    const unsigned short* __restrict__ qd,
    const unsigned short* __restrict__ kd,
    const unsigned short* __restrict__ vtd,
    unsigned short* __restrict__ ao) {
  const int lane = threadIdx.x & 31;
  const int wave = blockIdx.x * (blockDim.x >> 5) + (threadIdx.x >> 5);
  const int qtile = wave & (TT / 16 - 1);   // 128
  const int bh = wave >> 7;                 // 0..63
  const int b = bh >> 4, h = bh & 15;
  const unsigned short* qp = qd + ((size_t)bh * TT + qtile * 16) * HD;
  const unsigned short* kp = kd + (size_t)bh * TT * HD;
  const unsigned short* vp = vtd + (size_t)bh * HD * TT;

  const u16x16 qf0 = load_B(qp, HD);        // B of S^T gemm: hd 0..31
  const u16x16 qf1 = load_B(qp + 32, HD);   // hd 32..63

  f32x8 acc0 = {}, acc1 = {}, acc2 = {}, acc3 = {};
  float m = -1e30f, l = 0.f;
  const int qb16 = (lane >> 4) << 3;

  for (int j = 0; j < TT; j += 32) {
    // K fragments for this chunk, then V fragments: the score WMMAs only wait
    // for the K group (loadcnt<=8) while V stays in flight under the softmax.
    u16x16 ka0 = load_A(kp + (size_t)j * HD, HD);
    u16x16 ka1 = load_A(kp + (size_t)j * HD + 32, HD);
    u16x16 kc0 = load_A(kp + (size_t)(j + 16) * HD, HD);
    u16x16 kc1 = load_A(kp + (size_t)(j + 16) * HD + 32, HD);
    u16x16 vf0 = load_B(vp + (size_t)0 * 16 * TT + j, TT);
    u16x16 vf1 = load_B(vp + (size_t)1 * 16 * TT + j, TT);
    u16x16 vf2 = load_B(vp + (size_t)2 * 16 * TT + j, TT);
    u16x16 vf3 = load_B(vp + (size_t)3 * 16 * TT + j, TT);

    f32x8 s0 = {}, s1 = {};
    s0 = wmma_bf16(ka0, qf0, s0);
    s0 = wmma_bf16(ka1, qf1, s0);
    s1 = wmma_bf16(kc0, qf0, s1);
    s1 = wmma_bf16(kc1, qf1, s1);

    float cmax = -1e30f;
#pragma unroll
    for (int r = 0; r < 8; ++r) cmax = fmaxf(cmax, fmaxf(s0[r], s1[r]));
    cmax = fmaxf(cmax, __shfl_xor(cmax, 16, 32));
    const float mn = fmaxf(m, cmax);
    const float alpha = __expf(m - mn);
    m = mn;

    u16x16 pa;
    float ls = 0.f;
#pragma unroll
    for (int r = 0; r < 8; ++r) {
      const float e0 = __expf(s0[r] - mn);
      const float e1 = __expf(s1[r] - mn);
      ls += e0 + e1;
      pa[r]     = f2bf(e0);   // keys j+{0..7 | 8..15}    -> A elements 0..7
      pa[r + 8] = f2bf(e1);   // keys j+16+{0..7 | 8..15} -> A elements 8..15
    }
    ls += __shfl_xor(ls, 16, 32);
    l = l * alpha + ls;

#pragma unroll
    for (int r = 0; r < 8; ++r) {
      const float ar = __shfl(alpha, qb16 + r, 32);
      acc0[r] *= ar; acc1[r] *= ar; acc2[r] *= ar; acc3[r] *= ar;
    }

    acc0 = wmma_bf16(pa, vf0, acc0);
    acc1 = wmma_bf16(pa, vf1, acc1);
    acc2 = wmma_bf16(pa, vf2, acc2);
    acc3 = wmma_bf16(pa, vf3, acc3);
  }

  const int col = lane & 15;
#pragma unroll
  for (int r = 0; r < 8; ++r) {
    const float li = 1.0f / __shfl(l, qb16 + r, 32);
    const int t = qtile * 16 + qb16 + r;
    unsigned short* o = ao + (size_t)(b * TT + t) * DD + h * HD;
    o[0 * 16 + col] = f2bf(acc0[r] * li);
    o[1 * 16 + col] = f2bf(acc1[r] * li);
    o[2 * 16 + col] = f2bf(acc2[r] * li);
    o[3 * 16 + col] = f2bf(acc3[r] * li);
  }
}

// ---------------------------------------------------------------------------
// out = attn_out @ W_proj^T  (fp32 output), 16x64 wave tile, ping-pong buffered
// ---------------------------------------------------------------------------
__global__ void __launch_bounds__(256) proj_gemm(
    const unsigned short* __restrict__ ao,
    const unsigned short* __restrict__ wp,
    float* __restrict__ out) {
  const int lane = threadIdx.x & 31;
  const int wave = blockIdx.x * (blockDim.x >> 5) + (threadIdx.x >> 5);
  const int nstrip = wave & 15;   // 16 strips of 64 over 1024
  const int mtile  = wave >> 4;   // 512 tiles of 16 rows
  const unsigned short* ap = ao + (size_t)mtile * 16 * DD;
  const unsigned short* bp = wp + (size_t)(nstrip * 64) * DD;

  f32x8 acc[4] = {};
  u16x16 a0, a1, b0[4], b1[4];

  a0 = load_A(ap, DD);
#pragma unroll
  for (int jt = 0; jt < 4; ++jt) b0[jt] = load_B(bp + (size_t)jt * 16 * DD, DD);

  for (int kk = 0; kk < DD - 64; kk += 64) {
    a1 = load_A(ap + kk + 32, DD);
#pragma unroll
    for (int jt = 0; jt < 4; ++jt) b1[jt] = load_B(bp + (size_t)jt * 16 * DD + kk + 32, DD);
#pragma unroll
    for (int jt = 0; jt < 4; ++jt) acc[jt] = wmma_bf16(a0, b0[jt], acc[jt]);

    a0 = load_A(ap + kk + 64, DD);
#pragma unroll
    for (int jt = 0; jt < 4; ++jt) b0[jt] = load_B(bp + (size_t)jt * 16 * DD + kk + 64, DD);
#pragma unroll
    for (int jt = 0; jt < 4; ++jt) acc[jt] = wmma_bf16(a1, b1[jt], acc[jt]);
  }
  a1 = load_A(ap + DD - 32, DD);
#pragma unroll
  for (int jt = 0; jt < 4; ++jt) b1[jt] = load_B(bp + (size_t)jt * 16 * DD + DD - 32, DD);
#pragma unroll
  for (int jt = 0; jt < 4; ++jt) acc[jt] = wmma_bf16(a0, b0[jt], acc[jt]);
#pragma unroll
  for (int jt = 0; jt < 4; ++jt) acc[jt] = wmma_bf16(a1, b1[jt], acc[jt]);

  const int col  = lane & 15;
  const int qb16 = (lane >> 4) << 3;
#pragma unroll
  for (int jt = 0; jt < 4; ++jt) {
#pragma unroll
    for (int r = 0; r < 8; ++r) {
      const int grow = mtile * 16 + qb16 + r;
      out[(size_t)grow * DD + nstrip * 64 + jt * 16 + col] = acc[jt][r];
    }
  }
}

extern "C" void kernel_launch(void* const* d_in, const int* in_sizes, int n_in,
                              void* d_out, int out_size, void* d_ws, size_t ws_size,
                              hipStream_t stream) {
  (void)in_sizes; (void)n_in; (void)out_size; (void)ws_size;
  const float* x     = (const float*)d_in[0];
  const float* wqkv  = (const float*)d_in[1];
  const float* wproj = (const float*)d_in[2];
  float* out = (float*)d_out;

  char* ws = (char*)d_ws;
  unsigned short* xb     = (unsigned short*)(ws + 0);           // 16 MiB
  unsigned short* wqkvb  = (unsigned short*)(ws + 16777216);    //  6 MiB
  unsigned short* wprojb = (unsigned short*)(ws + 23068672);    //  2 MiB
  unsigned short* qd     = (unsigned short*)(ws + 25165824);    // 16 MiB
  unsigned short* kd     = (unsigned short*)(ws + 41943040);    // 16 MiB
  unsigned short* vtd    = (unsigned short*)(ws + 58720256);    // 16 MiB
  unsigned short* ao     = (unsigned short*)(ws + 75497472);    // 16 MiB

  const int NX  = BB * TT * DD;
  const int NWQ = 3 * DD * DD;
  const int NWP = DD * DD;
  cvt_bf16<<<(NX  + 255) / 256, 256, 0, stream>>>(x, xb, NX);
  cvt_bf16<<<(NWQ + 255) / 256, 256, 0, stream>>>(wqkv, wqkvb, NWQ);
  cvt_bf16<<<(NWP + 255) / 256, 256, 0, stream>>>(wproj, wprojb, NWP);

  // 512 mtiles * 48 nstrips = 24576 waves, 8 waves/block
  qkv_gemm<<<3072, 256, 0, stream>>>(xb, wqkvb, qd, kd, vtd);
  // 64 (b,h) * 128 qtiles = 8192 waves
  attn_kernel<<<1024, 256, 0, stream>>>(qd, kd, vtd, ao);
  // 512 mtiles * 16 nstrips = 8192 waves
  proj_gemm<<<1024, 256, 0, stream>>>(ao, wprojb, out);
}
